// Het_GIN_58007828300388
// MI455X (gfx1250) — compile-verified, matrix operand order checked by reading, not run on hardware
//
#include <hip/hip_runtime.h>

#define DIM   256
#define LDSS  264            // padded LDS row stride (bf16 elems): 264*2B = 528B -> bank-conflict-free
#define NA    50000
#define NM    10000
#define NG    500
#define LAY   3

typedef __attribute__((ext_vector_type(16))) __bf16 v16bf;
typedef __attribute__((ext_vector_type(8)))  __bf16 v8bf;
typedef __attribute__((ext_vector_type(8)))  float  v8f;

// ---------------------------------------------------------------------------
// Weight prep: W (nmats x 256 x 256, [k][n], f32)  ->  WT ([n][k], bf16)
// ---------------------------------------------------------------------------
__global__ __launch_bounds__(256)
void prep_weights_kernel(const float* __restrict__ W, __bf16* __restrict__ WT, int nmats)
{
    long i = (long)blockIdx.x * blockDim.x + threadIdx.x;
    long total = (long)nmats << 16;
    if (i >= total) return;
    long mat = i >> 16;
    int  kn  = (int)(i & 65535);
    int  k   = kn >> 8;
    int  n   = kn & 255;
    WT[(mat << 16) + ((long)n << 8) + k] = (__bf16)W[i];
}

// ---------------------------------------------------------------------------
// Edge messages: agg[dst] += relu(x_src[src] + attr)   (one thread = 4 dims)
// ---------------------------------------------------------------------------
__global__ __launch_bounds__(256)
void edge_msg_kernel(const float* __restrict__ xsrc, const float* __restrict__ attr,
                     const int* __restrict__ src, const int* __restrict__ dst,
                     float* __restrict__ agg, int E)
{
    long tid = (long)blockIdx.x * blockDim.x + threadIdx.x;
    long e   = tid >> 6;
    if (e >= E) return;
    int  q = (int)(tid & 63) * 4;
    int  s = src[e];
    int  d = dst[e];
    float4 a = *(const float4*)(attr + e * (long)DIM + q);
    float4 x = *(const float4*)(xsrc + (long)s * DIM + q);
    float4 m;
    m.x = fmaxf(a.x + x.x, 0.f);
    m.y = fmaxf(a.y + x.y, 0.f);
    m.z = fmaxf(a.z + x.z, 0.f);
    m.w = fmaxf(a.w + x.w, 0.f);
    float* o = agg + (long)d * DIM + q;
    unsafeAtomicAdd(o + 0, m.x);
    unsafeAtomicAdd(o + 1, m.y);
    unsafeAtomicAdd(o + 2, m.z);
    unsafeAtomicAdd(o + 3, m.w);
}

// ---------------------------------------------------------------------------
// Fused dual-conv node MLP (bf16 WMMA, f32 accumulate)
//   out = relu((aggA+x)@W1A+b1A)@W2A + b2A + relu((aggB+x)@W1B+b1B)@W2B + b2B
// One block = 16 node rows x 256 cols.  8 wave32 / block, 2 N-tiles per wave.
// ---------------------------------------------------------------------------
__device__ __forceinline__ v16bf load_frag(const __bf16* p)
{
    v8bf lo = *(const v8bf*)p;
    v8bf hi = *(const v8bf*)(p + 16);
    return __builtin_shufflevector(lo, hi, 0,1,2,3,4,5,6,7,8,9,10,11,12,13,14,15);
}

__global__ __launch_bounds__(256)
void node_mlp_kernel(const float* __restrict__ aggA, const float* __restrict__ aggB,
                     const float* __restrict__ xdst,
                     const __bf16* __restrict__ W1A, const float* __restrict__ b1A,
                     const __bf16* __restrict__ W2A, const float* __restrict__ b2A,
                     const __bf16* __restrict__ W1B, const float* __restrict__ b1B,
                     const __bf16* __restrict__ W2B, const float* __restrict__ b2B,
                     float* __restrict__ out)
{
    __shared__ __bf16 hA[16 * LDSS];   // conv input tile (bf16)
    __shared__ __bf16 hH[16 * LDSS];   // hidden tile (bf16)

    const int  tid  = threadIdx.x;
    const int  wave = tid >> 5;        // 0..7
    const int  lane = tid & 31;
    const int  half = lane >> 4;       // 0/1
    const int  lrow = lane & 15;       // A: row   B/C: column-in-tile
    const long rowbase = (long)blockIdx.x * 16;

    v8f accOut0 = {};                  // N-tile 2*wave
    v8f accOut1 = {};                  // N-tile 2*wave+1

    #pragma unroll 1
    for (int pass = 0; pass < 2; ++pass) {
        const float*  agg = pass ? aggB : aggA;
        const __bf16* W1  = pass ? W1B  : W1A;
        const float*  b1  = pass ? b1B  : b1A;
        const __bf16* W2  = pass ? W2B  : W2A;

        // ---- stage 1: hA = bf16(agg + x) -------------------------------
        {
            int r  = tid >> 4;                 // 0..15
            int c0 = (tid & 15) * 16;          // 16 cols per thread
            const float* pa = agg  + (rowbase + r) * DIM + c0;
            const float* px = xdst + (rowbase + r) * DIM + c0;
            __bf16* ph = hA + r * LDSS + c0;
            #pragma unroll
            for (int j = 0; j < 4; ++j) {
                float4 a = ((const float4*)pa)[j];
                float4 x = ((const float4*)px)[j];
                ph[j*4+0] = (__bf16)(a.x + x.x);
                ph[j*4+1] = (__bf16)(a.y + x.y);
                ph[j*4+2] = (__bf16)(a.z + x.z);
                ph[j*4+3] = (__bf16)(a.w + x.w);
            }
        }
        __syncthreads();

        // ---- stage 2: hidden = relu(hA @ W1 + b1) -> hH ----------------
        #pragma unroll
        for (int t = 0; t < 2; ++t) {
            const int nt = wave * 2 + t;
            v8f acc = {};
            #pragma unroll
            for (int kk = 0; kk < 8; ++kk) {
                v16bf afrag = load_frag(hA + lrow * LDSS + kk * 32 + half * 8);
                v16bf bfrag = load_frag(W1 + ((long)(nt * 16 + lrow) << 8) + kk * 32 + half * 8);
                acc = __builtin_amdgcn_wmma_f32_16x16x32_bf16(
                          false, afrag, false, bfrag, (short)0, acc, false, false);
            }
            const int n   = nt * 16 + lrow;
            const float b = b1[n];
            #pragma unroll
            for (int v = 0; v < 8; ++v) {
                int m = v + half * 8;
                hH[m * LDSS + n] = (__bf16)fmaxf(acc[v] + b, 0.f);
            }
        }
        __syncthreads();

        // ---- stage 3: accOut += hH @ W2 --------------------------------
        #pragma unroll
        for (int t = 0; t < 2; ++t) {
            const int nt = wave * 2 + t;
            v8f acc = t ? accOut1 : accOut0;
            #pragma unroll
            for (int kk = 0; kk < 8; ++kk) {
                v16bf afrag = load_frag(hH + lrow * LDSS + kk * 32 + half * 8);
                v16bf bfrag = load_frag(W2 + ((long)(nt * 16 + lrow) << 8) + kk * 32 + half * 8);
                acc = __builtin_amdgcn_wmma_f32_16x16x32_bf16(
                          false, afrag, false, bfrag, (short)0, acc, false, false);
            }
            if (t) accOut1 = acc; else accOut0 = acc;
        }
        __syncthreads();   // hA/hH reused next pass
    }

    // ---- epilogue: + b2A + b2B, store f32 ------------------------------
    #pragma unroll
    for (int t = 0; t < 2; ++t) {
        const int nt = wave * 2 + t;
        const int n  = nt * 16 + lrow;
        const float b = b2A[n] + b2B[n];
        v8f acc = t ? accOut1 : accOut0;
        #pragma unroll
        for (int v = 0; v < 8; ++v) {
            int m = v + half * 8;
            out[(rowbase + m) * DIM + n] = acc[v] + b;
        }
    }
}

// ---------------------------------------------------------------------------
// Global-add-pool into out[:, layer*256 : layer*256+256]
// ---------------------------------------------------------------------------
__global__ __launch_bounds__(256)
void pool_kernel(const float* __restrict__ x, const int* __restrict__ batch,
                 float* __restrict__ out, int n, int layer)
{
    long tid  = (long)blockIdx.x * blockDim.x + threadIdx.x;
    long node = tid >> 6;
    if (node >= n) return;
    int q = (int)(tid & 63) * 4;
    int g = batch[node];
    float4 v = *(const float4*)(x + node * (long)DIM + q);
    float* o = out + (long)g * (LAY * DIM) + layer * DIM + q;
    unsafeAtomicAdd(o + 0, v.x);
    unsafeAtomicAdd(o + 1, v.y);
    unsafeAtomicAdd(o + 2, v.z);
    unsafeAtomicAdd(o + 3, v.w);
}

// ---------------------------------------------------------------------------
extern "C" void kernel_launch(void* const* d_in, const int* in_sizes, int n_in,
                              void* d_out, int out_size, void* d_ws, size_t ws_size,
                              hipStream_t stream)
{
    const float* x_atom  = (const float*)d_in[0];
    const float* x_motif = (const float*)d_in[1];
    const float* attr_aa = (const float*)d_in[2];
    const float* attr_am = (const float*)d_in[3];
    const float* attr_ma = (const float*)d_in[4];
    const float* attr_mm = (const float*)d_in[5];
    const float* W1      = (const float*)d_in[6];
    const float* b1      = (const float*)d_in[7];
    const float* W2      = (const float*)d_in[8];
    const float* b2      = (const float*)d_in[9];
    const int* ei_aa     = (const int*)d_in[10];
    const int* ei_mm     = (const int*)d_in[11];
    const int* src_am    = (const int*)d_in[12];
    const int* dst_am    = (const int*)d_in[13];
    const int* src_ma    = (const int*)d_in[14];
    const int* dst_ma    = (const int*)d_in[15];
    const int* batch_a   = (const int*)d_in[16];
    const int* batch_m   = (const int*)d_in[17];
    float* outp = (float*)d_out;

    // ---- workspace carve-up (bytes) ----
    char* ws = (char*)d_ws;
    const size_t SZ_NA = (size_t)NA * DIM * sizeof(float);   // 51.2 MB
    const size_t SZ_NM = (size_t)NM * DIM * sizeof(float);   // 10.24 MB
    const size_t SZ_WT = (size_t)12 * 65536 * sizeof(__bf16);
    float*  agg_aa = (float*)(ws);                 ws += SZ_NA;
    float*  agg_ma = (float*)(ws);                 ws += SZ_NA;
    float*  agg_am = (float*)(ws);                 ws += SZ_NM;
    float*  agg_mm = (float*)(ws);                 ws += SZ_NM;
    float*  xa_buf[2]; xa_buf[0] = (float*)(ws);   ws += SZ_NA;
                       xa_buf[1] = (float*)(ws);   ws += SZ_NA;
    float*  xm_buf[2]; xm_buf[0] = (float*)(ws);   ws += SZ_NM;
                       xm_buf[1] = (float*)(ws);   ws += SZ_NM;
    __bf16* W1T = (__bf16*)(ws);                   ws += SZ_WT;
    __bf16* W2T = (__bf16*)(ws);                   ws += SZ_WT;

    // ---- zero output, prep bf16 transposed weights ----
    hipMemsetAsync(d_out, 0, (size_t)NG * LAY * DIM * sizeof(float), stream);
    {
        int total = 12 * 65536;
        prep_weights_kernel<<<(total + 255) / 256, 256, 0, stream>>>(W1, W1T, 12);
        prep_weights_kernel<<<(total + 255) / 256, 256, 0, stream>>>(W2, W2T, 12);
    }

    for (int l = 0; l < LAY; ++l) {
        const float* xa_in = (l == 0) ? x_atom  : xa_buf[(l + 1) & 1];
        const float* xm_in = (l == 0) ? x_motif : xm_buf[(l + 1) & 1];
        float* xa_out = xa_buf[l & 1];
        float* xm_out = xm_buf[l & 1];

        hipMemsetAsync(agg_aa, 0, SZ_NA, stream);
        hipMemsetAsync(agg_ma, 0, SZ_NA, stream);
        hipMemsetAsync(agg_am, 0, SZ_NM, stream);
        hipMemsetAsync(agg_mm, 0, SZ_NM, stream);

        // messages (1 thread = 4 dims => E/4 blocks of 256)
        edge_msg_kernel<<<400000 / 4, 256, 0, stream>>>(xa_in, attr_aa, ei_aa, ei_aa + 400000, agg_aa, 400000);
        edge_msg_kernel<<<150000 / 4, 256, 0, stream>>>(xm_in, attr_ma, src_ma, dst_ma,        agg_ma, 150000);
        edge_msg_kernel<<<150000 / 4, 256, 0, stream>>>(xa_in, attr_am, src_am, dst_am,        agg_am, 150000);
        edge_msg_kernel<<<100000 / 4, 256, 0, stream>>>(xm_in, attr_mm, ei_mm, ei_mm + 100000, agg_mm, 100000);

        // per-(layer,type) parameter slices: type 0=aa 1=am 2=ma 3=mm
        const long m0 = (long)(l * 4 + 0) << 16, m1 = (long)(l * 4 + 1) << 16;
        const long m2 = (long)(l * 4 + 2) << 16, m3 = (long)(l * 4 + 3) << 16;
        const long v0 = (long)(l * 4 + 0) << 8,  v1 = (long)(l * 4 + 1) << 8;
        const long v2 = (long)(l * 4 + 2) << 8,  v3 = (long)(l * 4 + 3) << 8;

        // atoms: MLP_t0(agg_aa + xa) + MLP_t2(agg_ma + xa)
        node_mlp_kernel<<<NA / 16, 256, 0, stream>>>(
            agg_aa, agg_ma, xa_in,
            W1T + m0, b1 + v0, W2T + m0, b2 + v0,
            W1T + m2, b1 + v2, W2T + m2, b2 + v2,
            xa_out);

        // motifs: MLP_t1(agg_am + xm) + MLP_t3(agg_mm + xm)
        node_mlp_kernel<<<NM / 16, 256, 0, stream>>>(
            agg_am, agg_mm, xm_in,
            W1T + m1, b1 + v1, W2T + m1, b2 + v1,
            W1T + m3, b1 + v3, W2T + m3, b2 + v3,
            xm_out);

        // pooled[g, l*256:+256] += sum(xa) + sum(xm)
        pool_kernel<<<NA / 4, 256, 0, stream>>>(xa_out, batch_a, outp, NA, l);
        pool_kernel<<<NM / 4, 256, 0, stream>>>(xm_out, batch_m, outp, NM, l);
    }
}
